// MyAlbertForMaskedLM_28638841930024
// MI455X (gfx1250) — matrix operand order repaired and checked by reading, not compiled, not run on hardware
//
#include <hip/hip_runtime.h>
#include <hip/hip_bf16.h>

// ---------------- model dims ----------------
#define BB 64
#define LL 64
#define VV 30000
#define EE 128
#define HH 768
#define NHH 12
#define DHH 64
#define FFD 3072
#define NLAYER 12
#define KK 16
#define ROWS (BB * LL)   // 4096 token rows
#define MROWS (KK * BB)  // 1024 masked slots

typedef __attribute__((ext_vector_type(16))) __bf16 v16bf;
typedef __attribute__((ext_vector_type(8)))  float  v8f;

__device__ __forceinline__ v8f wmma_bf16(v16bf a, v16bf b, v8f c) {
  // D = A(16x32 bf16) * B(32x16 bf16) + C(16x16 f32)
  return __builtin_amdgcn_wmma_f32_16x16x32_bf16(false, a, false, b, (short)0, c,
                                                 false, false);
}

__device__ __forceinline__ float gelu_f(float x) {
  float x3 = x * x * x;
  return 0.5f * x * (1.0f + tanhf(0.7978845608028654f * (x + 0.044715f * x3)));
}

union Frag { uint4 u[2]; v16bf v; };
union Pack8 { uint4 u; __bf16 h[8]; };

// ---------------- generic bf16 WMMA GEMM: C = act(A @ Bt^T + bias) -----------
// A:  [M, Kd] bf16 row-major (M % 128 == 0, Kd % 32 == 0)
// Bt: [N, Kd] bf16 row-major (TRANSPOSED weight; N may be ragged)
// Block tile 128x64, 8 waves, each wave 32x32 (4 WMMA per K-step).
__global__ __launch_bounds__(256) void gemm_bf16_kernel(
    const __bf16* __restrict__ A, const __bf16* __restrict__ Bt,
    const float* __restrict__ bias, float* __restrict__ Cf,
    __bf16* __restrict__ Cb, int M, int N, int Kd, int act) {
  __shared__ __bf16 sA[128][40];  // [m][k]  80B row stride (16B aligned)
  __shared__ __bf16 sB[64][40];   // [n][k]  80B row stride (16B aligned)

  const int tid  = threadIdx.x;
  const int lane = tid & 31;
  const int wid  = tid >> 5;        // 8 waves
  const int m0   = blockIdx.y * 128;
  const int n0   = blockIdx.x * 64;
  const int wm   = wid >> 1;        // 0..3 : 32-row band
  const int wn   = wid & 1;         // 0..1 : 32-col band

  v8f acc00 = {}, acc01 = {}, acc10 = {}, acc11 = {};

  // staging coords
  const int ar = tid >> 1;          // 0..127
  const int ac = (tid & 1) * 16;    // 0,16
  const int br = tid >> 2;          // 0..63
  const int bc = (tid & 3) * 8;     // 0,8,16,24
  const bool brow_ok = (n0 + br) < N;

  const int kh = (lane >> 4) & 1;   // A half-select
  const int ko = (lane >> 4) * 16;  // B K-offset
  const int ml = lane & 15;

  for (int k0 = 0; k0 < Kd; k0 += 32) {
    // stage A tile 128x32: 2 x b128 per thread
    *(uint4*)&sA[ar][ac] =
        *(const uint4*)(A + (size_t)(m0 + ar) * Kd + k0 + ac);
    *(uint4*)&sA[ar][ac + 8] =
        *(const uint4*)(A + (size_t)(m0 + ar) * Kd + k0 + ac + 8);
    // stage Bt tile 64x32: 1 x b128 per thread, per-row ragged guard
    if (brow_ok) {
      *(uint4*)&sB[br][bc] =
          *(const uint4*)(Bt + (size_t)(n0 + br) * Kd + k0 + bc);
    } else {
      uint4 z = {0, 0, 0, 0};
      *(uint4*)&sB[br][bc] = z;
    }
    if (k0 + 32 < Kd) {  // global_prefetch next K-slab
      __builtin_prefetch(A + (size_t)(m0 + ar) * Kd + k0 + 32 + ac, 0, 1);
      __builtin_prefetch(Bt + (size_t)(n0 + br) * Kd + k0 + 32 + bc, 0, 1);
    }
    __syncthreads();

    // A fragments: lane half h holds K 16h..16h+15 (same order as B)
    Frag a0, a1, b0, b1;
    const int ma = wm * 32 + ml;
    a0.u[0] = *(const uint4*)&sA[ma][ko];
    a0.u[1] = *(const uint4*)&sA[ma][ko + 8];
    a1.u[0] = *(const uint4*)&sA[ma + 16][ko];
    a1.u[1] = *(const uint4*)&sA[ma + 16][ko + 8];
    // B fragments: lane n holds K ko..ko+15 of column n == row of Bt
    const int nb = wn * 32 + ml;
    b0.u[0] = *(const uint4*)&sB[nb][ko];
    b0.u[1] = *(const uint4*)&sB[nb][ko + 8];
    b1.u[0] = *(const uint4*)&sB[nb + 16][ko];
    b1.u[1] = *(const uint4*)&sB[nb + 16][ko + 8];

    acc00 = wmma_bf16(a0.v, b0.v, acc00);
    acc01 = wmma_bf16(a0.v, b1.v, acc01);
    acc10 = wmma_bf16(a1.v, b0.v, acc10);
    acc11 = wmma_bf16(a1.v, b1.v, acc11);
    __syncthreads();
  }

  // epilogue: C/D layout => VGPR r holds (M=(lane>=16?8:0)+r, N=lane&15)
  const int rbase = m0 + wm * 32 + (lane >> 4) * 8;
  const int cbase = n0 + wn * 32 + ml;
#pragma unroll
  for (int mt = 0; mt < 2; ++mt) {
#pragma unroll
    for (int nt = 0; nt < 2; ++nt) {
      const v8f& acc = mt == 0 ? (nt == 0 ? acc00 : acc01)
                               : (nt == 0 ? acc10 : acc11);
      const int col = cbase + nt * 16;
      if (col < N) {
        const float bv = bias ? bias[col] : 0.0f;
#pragma unroll
        for (int r = 0; r < 8; ++r) {
          const int row = rbase + mt * 16 + r;
          float v = acc[r] + bv;
          if (act == 1) v = gelu_f(v);
          if (Cf) Cf[(size_t)row * N + col] = v;
          if (Cb) Cb[(size_t)row * N + col] = (__bf16)v;
        }
      }
    }
  }
}

// ---------------- per-(head,batch) attention: softmax(QK^T*s + ext) V --------
__global__ __launch_bounds__(256) void attention_kernel(
    const __bf16* __restrict__ Q, const __bf16* __restrict__ Kt,
    const __bf16* __restrict__ Vt, const int* __restrict__ am,
    __bf16* __restrict__ Ctx) {
  __shared__ __bf16 sQ[64][72];   // [q][d]   row-major (A operand)
  __shared__ __bf16 sK[64][72];   // [key][d] row-major (B operand of QK^T)
  __shared__ __bf16 sVt[64][72];  // [d][key] transposed (B operand of PV)
  __shared__ float  sS[64][64];
  __shared__ __bf16 sP[64][72];   // [q][key] row-major (A operand of PV)

  const int h = blockIdx.x;   // head
  const int b = blockIdx.y;   // batch
  const int tid  = threadIdx.x;
  const int lane = tid & 31;
  const int wid  = tid >> 5;
  const int wm   = wid >> 1;  // 0..3
  const int wn   = wid & 1;   // 0..1

  // staging: b128 global loads; V transposed into LDS
  for (int i = tid; i < 512; i += 256) {
    const int r = i >> 3;          // 0..63
    const int cb = (i & 7) * 8;    // 0..56
    const size_t g = (size_t)(b * 64 + r) * HH + h * 64 + cb;
    *(uint4*)&sQ[r][cb] = *(const uint4*)(Q + g);
    *(uint4*)&sK[r][cb] = *(const uint4*)(Kt + g);
    Pack8 pv;
    pv.u = *(const uint4*)(Vt + g);
#pragma unroll
    for (int j = 0; j < 8; ++j) sVt[cb + j][r] = pv.h[j];
  }
  __syncthreads();

  const int ml = lane & 15;
  const int kh = (lane >> 4) & 1;
  const int ko = (lane >> 4) * 16;
  const int m  = wm * 16 + ml;

  // S = Q @ K^T over DH=64 (two K-steps of 32)
  v8f a0 = {}, a1 = {};
#pragma unroll
  for (int ks = 0; ks < 2; ++ks) {
    Frag af, b0, b1;
    af.u[0] = *(const uint4*)&sQ[m][ks * 32 + ko];
    af.u[1] = *(const uint4*)&sQ[m][ks * 32 + ko + 8];
    b0.u[0] = *(const uint4*)&sK[wn * 32 + ml][ks * 32 + ko];
    b0.u[1] = *(const uint4*)&sK[wn * 32 + ml][ks * 32 + ko + 8];
    b1.u[0] = *(const uint4*)&sK[wn * 32 + 16 + ml][ks * 32 + ko];
    b1.u[1] = *(const uint4*)&sK[wn * 32 + 16 + ml][ks * 32 + ko + 8];
    a0 = wmma_bf16(af.v, b0.v, a0);
    a1 = wmma_bf16(af.v, b1.v, a1);
  }
  {
    const float scale = 0.125f;  // 1/sqrt(64)
    const int rb = wm * 16 + (lane >> 4) * 8;
    const int c0 = wn * 32 + ml;
    const float e0 = (1.0f - (float)am[b * 64 + c0]) * -10000.0f;
    const float e1 = (1.0f - (float)am[b * 64 + c0 + 16]) * -10000.0f;
#pragma unroll
    for (int r = 0; r < 8; ++r) {
      sS[rb + r][c0]      = a0[r] * scale + e0;
      sS[rb + r][c0 + 16] = a1[r] * scale + e1;
    }
  }
  __syncthreads();

  // fp32 row softmax (64 rows)
  if (tid < 64) {
    float mx = -1e30f;
    for (int c = 0; c < 64; ++c) mx = fmaxf(mx, sS[tid][c]);
    float sum = 0.0f;
    for (int c = 0; c < 64; ++c) {
      float e = expf(sS[tid][c] - mx);
      sS[tid][c] = e;
      sum += e;
    }
    float inv = 1.0f / sum;
    for (int c = 0; c < 64; ++c) sP[tid][c] = (__bf16)(sS[tid][c] * inv);
  }
  __syncthreads();

  // Ctx = P @ V   (B operand from sVt => contiguous b128 frags)
  v8f c0v = {}, c1v = {};
#pragma unroll
  for (int ks = 0; ks < 2; ++ks) {
    Frag af, b0, b1;
    af.u[0] = *(const uint4*)&sP[m][ks * 32 + ko];
    af.u[1] = *(const uint4*)&sP[m][ks * 32 + ko + 8];
    b0.u[0] = *(const uint4*)&sVt[wn * 32 + ml][ks * 32 + ko];
    b0.u[1] = *(const uint4*)&sVt[wn * 32 + ml][ks * 32 + ko + 8];
    b1.u[0] = *(const uint4*)&sVt[wn * 32 + 16 + ml][ks * 32 + ko];
    b1.u[1] = *(const uint4*)&sVt[wn * 32 + 16 + ml][ks * 32 + ko + 8];
    c0v = wmma_bf16(af.v, b0.v, c0v);
    c1v = wmma_bf16(af.v, b1.v, c1v);
  }
  {
    const int rb = wm * 16 + (lane >> 4) * 8;
    const int cc = wn * 32 + ml;
#pragma unroll
    for (int r = 0; r < 8; ++r) {
      const size_t g = (size_t)(b * 64 + rb + r) * HH + h * 64;
      Ctx[g + cc]      = (__bf16)c0v[r];
      Ctx[g + cc + 16] = (__bf16)c1v[r];
    }
  }
}

// ---------------- residual + LayerNorm (fp32), emits fp32 and/or bf16 --------
__global__ __launch_bounds__(256) void add_ln_kernel(
    const float* __restrict__ base, const float* __restrict__ delta,
    const float* __restrict__ g, const float* __restrict__ bta,
    float* __restrict__ outF, __bf16* __restrict__ outB, int D) {
  __shared__ float red[256];
  const int row = blockIdx.x;
  const int tid = threadIdx.x;
  float x[4];
  int cnt = 0;
  float s = 0.0f;
  for (int j = tid, i = 0; j < D; j += 256, ++i) {
    float v = base[(size_t)row * D + j] +
              (delta ? delta[(size_t)row * D + j] : 0.0f);
    x[i] = v;
    s += v;
    cnt = i + 1;
  }
  red[tid] = s;
  __syncthreads();
  for (int st = 128; st > 0; st >>= 1) {
    if (tid < st) red[tid] += red[tid + st];
    __syncthreads();
  }
  const float mean = red[0] / (float)D;
  __syncthreads();
  float s2 = 0.0f;
  for (int i = 0; i < cnt; ++i) {
    float d = x[i] - mean;
    s2 += d * d;
  }
  red[tid] = s2;
  __syncthreads();
  for (int st = 128; st > 0; st >>= 1) {
    if (tid < st) red[tid] += red[tid + st];
    __syncthreads();
  }
  const float inv = rsqrtf(red[0] / (float)D + 1e-12f);
  for (int j = tid, i = 0; j < D; j += 256, ++i) {
    float v = (x[i] - mean) * inv * g[j] + bta[j];
    if (outF) outF[(size_t)row * D + j] = v;
    if (outB) outB[(size_t)row * D + j] = (__bf16)v;
  }
}

// ---------------- embedding lookup + LN -> bf16 [ROWS, EE] -------------------
__global__ __launch_bounds__(128) void embed_ln_kernel(
    const float* __restrict__ we, const float* __restrict__ pe,
    const float* __restrict__ te, const int* __restrict__ ids,
    const int* __restrict__ tt, const float* __restrict__ g,
    const float* __restrict__ bta, __bf16* __restrict__ out) {
  __shared__ float red[128];
  const int row = blockIdx.x;  // b*64 + t
  const int t = row & 63;
  const int e = threadIdx.x;   // 0..127
  const int id = ids[row];
  const int tti = tt[row];
  float v = we[(size_t)id * EE + e] + pe[t * EE + e] + te[tti * EE + e];
  red[e] = v;
  __syncthreads();
  for (int st = 64; st > 0; st >>= 1) {
    if (e < st) red[e] += red[e + st];
    __syncthreads();
  }
  const float mean = red[0] / (float)EE;
  __syncthreads();
  const float d = v - mean;
  red[e] = d * d;
  __syncthreads();
  for (int st = 64; st > 0; st >>= 1) {
    if (e < st) red[e] += red[e + st];
    __syncthreads();
  }
  const float inv = rsqrtf(red[0] / (float)EE + 1e-12f);
  out[(size_t)row * EE + e] = (__bf16)((v - mean) * inv * g[e] + bta[e]);
}

// ---------------- weight conversion kernels ----------------------------------
// plain fp32 -> bf16
__global__ void cvt_bf16_kernel(const float* __restrict__ src,
                                __bf16* __restrict__ dst, long n) {
  long i = (long)blockIdx.x * 256 + threadIdx.x;
  if (i < n) dst[i] = (__bf16)src[i];
}

// transpose-convert: dst[N,Kd] = src[Kd,N]^T (bf16), LDS-staged for coalescing
__global__ __launch_bounds__(256) void cvt_transpose_kernel(
    const float* __restrict__ src, __bf16* __restrict__ dst, int Kd, int N) {
  __shared__ float tile[32][33];
  const int kb = blockIdx.y * 32;
  const int nb = blockIdx.x * 32;
  const int tx = threadIdx.x & 31;
  const int ty = threadIdx.x >> 5;  // 0..7
  for (int j = ty; j < 32; j += 8) {
    const int k = kb + j, n = nb + tx;
    tile[j][tx] = (k < Kd && n < N) ? src[(size_t)k * N + n] : 0.0f;
  }
  __syncthreads();
  for (int j = ty; j < 32; j += 8) {
    const int n = nb + j, k = kb + tx;
    if (n < N && k < Kd) dst[(size_t)n * Kd + k] = (__bf16)tile[tx][j];
  }
}

// per-batch mask scan: excl[b,t] = #masked before t; slot (k,b) <- source row
__global__ void scan_slots_kernel(const int* __restrict__ mam,
                                  int* __restrict__ excl,
                                  int* __restrict__ slot_src) {
  int b = threadIdx.x;
  if (b >= BB) return;
  for (int k = 0; k < KK; ++k) slot_src[k * BB + b] = -1;
  int cnt = 0;
  for (int t = 0; t < LL; ++t) {
    excl[b * LL + t] = cnt;
    if (mam[b * LL + t]) {
      if (t >= 1 && cnt < KK) slot_src[cnt * BB + b] = b * LL + t;
      ++cnt;
    }
  }
}

__global__ __launch_bounds__(256) void gather_rows_kernel(
    const __bf16* __restrict__ hb, const int* __restrict__ slot_src,
    __bf16* __restrict__ hm) {
  const int s = blockIdx.x;
  const int src = slot_src[s];
  for (int j = threadIdx.x; j < HH; j += 256)
    hm[(size_t)s * HH + j] = (src >= 0) ? hb[(size_t)src * HH + j] : (__bf16)0.0f;
}

// per masked slot: max/argmax and sum-exp of (outputs + gumbel)/T
__global__ __launch_bounds__(256) void gumbel_stats_kernel(
    const float* __restrict__ scores, const float* __restrict__ rnd,
    const float* __restrict__ gum, const int* __restrict__ slot_src,
    float* __restrict__ statM, float* __restrict__ statS,
    int* __restrict__ statA) {
  __shared__ float sm[256];
  __shared__ int sa[256];
  const int s = blockIdx.x;
  const int tid = threadIdx.x;
  const bool wr = slot_src[s] >= 0;
  const float invT = 2.0f;  // 1/TEMP
  float mx = -1e30f;
  int arg = 0;
  for (int v = tid; v < VV; v += 256) {
    float val = wr ? scores[(size_t)s * VV + v] : rnd[(size_t)s * VV + v];
    float z = (val + gum[(size_t)s * VV + v]) * invT;
    if (z > mx) { mx = z; arg = v; }
  }
  sm[tid] = mx;
  sa[tid] = arg;
  __syncthreads();
  for (int st = 128; st > 0; st >>= 1) {
    if (tid < st && sm[tid + st] > sm[tid]) {
      sm[tid] = sm[tid + st];
      sa[tid] = sa[tid + st];
    }
    __syncthreads();
  }
  const float m = sm[0];
  const int a = sa[0];
  __syncthreads();
  float sum = 0.0f;
  for (int v = tid; v < VV; v += 256) {
    float val = wr ? scores[(size_t)s * VV + v] : rnd[(size_t)s * VV + v];
    float z = (val + gum[(size_t)s * VV + v]) * invT;
    sum += expf(z - m);
  }
  sm[tid] = sum;
  __syncthreads();
  for (int st = 128; st > 0; st >>= 1) {
    if (tid < st) sm[tid] += sm[tid + st];
    __syncthreads();
  }
  if (tid == 0) {
    statM[s] = m;
    statS[s] = sm[0];
    statA[s] = a;
  }
}

// final [B*L, V] output: one-hot for unmasked, straight-through gumbel for masked
__global__ __launch_bounds__(256) void scatter_out_kernel(
    const int* __restrict__ mam, const int* __restrict__ excl,
    const int* __restrict__ ids, const int* __restrict__ slot_src,
    const float* __restrict__ scores, const float* __restrict__ rnd,
    const float* __restrict__ gum, const float* __restrict__ statM,
    const float* __restrict__ statS, const int* __restrict__ statA,
    float* __restrict__ out) {
  const int row = blockIdx.y;               // b*64 + t
  const int v = blockIdx.x * 256 + threadIdx.x;
  if (v >= VV) return;
  const int b = row >> 6, t = row & 63;
  float o;
  if (mam[row] == 0) {
    // reference keeps transposed indexing: one_hot(input_ids[t, b])
    o = (v == ids[t * LL + b]) ? 1.0f : 0.0f;
  } else {
    int k = excl[row];
    if (k > KK - 1) k = KK - 1;
    const int s = k * BB + b;
    const bool wr = slot_src[s] >= 0;
    float val = wr ? scores[(size_t)s * VV + v] : rnd[(size_t)s * VV + v];
    float z = (val + gum[(size_t)s * VV + v]) * 2.0f;
    float y = expf(z - statM[s]) / statS[s];
    float yh = (v == statA[s]) ? 1.0f : 0.0f;
    o = (yh - y) + y;  // straight-through forward value
  }
  out[(size_t)row * VV + v] = o;
}

// ---------------- host orchestration ----------------------------------------
extern "C" void kernel_launch(void* const* d_in, const int* in_sizes, int n_in,
                              void* d_out, int out_size, void* d_ws,
                              size_t ws_size, hipStream_t stream) {
  (void)in_sizes; (void)n_in; (void)out_size; (void)ws_size;

  const float* word_emb = (const float*)d_in[0];
  const float* pos_emb  = (const float*)d_in[1];
  const float* tt_emb   = (const float*)d_in[2];
  const float* emb_ln_g = (const float*)d_in[3];
  const float* emb_ln_b = (const float*)d_in[4];
  const float* proj_w   = (const float*)d_in[5];
  const float* proj_bi  = (const float*)d_in[6];
  const float* wq = (const float*)d_in[7];   const float* bq = (const float*)d_in[8];
  const float* wk = (const float*)d_in[9];   const float* bk = (const float*)d_in[10];
  const float* wv = (const float*)d_in[11];  const float* bv = (const float*)d_in[12];
  const float* wo = (const float*)d_in[13];  const float* bo = (const float*)d_in[14];
  const float* ln1_g = (const float*)d_in[15]; const float* ln1_b = (const float*)d_in[16];
  const float* w1 = (const float*)d_in[17];  const float* b1 = (const float*)d_in[18];
  const float* w2 = (const float*)d_in[19];  const float* b2 = (const float*)d_in[20];
  const float* ln2_g = (const float*)d_in[21]; const float* ln2_b = (const float*)d_in[22];
  const float* mlm_w = (const float*)d_in[23]; const float* mlm_b = (const float*)d_in[24];
  const float* mlm_ln_g = (const float*)d_in[25]; const float* mlm_ln_b = (const float*)d_in[26];
  const float* dec_b = (const float*)d_in[27];
  const int* ids = (const int*)d_in[28];
  const int* mam = (const int*)d_in[29];
  const int* am  = (const int*)d_in[30];
  const int* tti = (const int*)d_in[31];
  const float* rnd = (const float*)d_in[32];
  const float* gum = (const float*)d_in[33];
  // d_in[34] = out_len (compile-time KK)

  char* ws = (char*)d_ws;
  size_t off = 0;
  auto alloc = [&](size_t bytes) -> void* {
    off = (off + 255) & ~(size_t)255;
    void* p = ws + off;
    off += bytes;
    return p;
  };

  // transposed bf16 weights: Wt[N][K]
  __bf16* projb = (__bf16*)alloc((size_t)HH * EE * 2);    // [H][E]
  __bf16* wqb   = (__bf16*)alloc((size_t)HH * HH * 2);
  __bf16* wkb   = (__bf16*)alloc((size_t)HH * HH * 2);
  __bf16* wvb   = (__bf16*)alloc((size_t)HH * HH * 2);
  __bf16* wob   = (__bf16*)alloc((size_t)HH * HH * 2);
  __bf16* w1b   = (__bf16*)alloc((size_t)FFD * HH * 2);   // [FF][H]
  __bf16* w2b   = (__bf16*)alloc((size_t)HH * FFD * 2);   // [H][FF]
  __bf16* mlmwb = (__bf16*)alloc((size_t)EE * HH * 2);    // [E][H]
  __bf16* wdecb = (__bf16*)alloc((size_t)VV * EE * 2);    // [V][E] = word_emb
  __bf16* embb  = (__bf16*)alloc((size_t)ROWS * EE * 2);
  float*  hF    = (float*)alloc((size_t)ROWS * HH * 4);
  __bf16* hB    = (__bf16*)alloc((size_t)ROWS * HH * 2);
  __bf16* qB    = (__bf16*)alloc((size_t)ROWS * HH * 2);
  __bf16* kB    = (__bf16*)alloc((size_t)ROWS * HH * 2);
  __bf16* vB    = (__bf16*)alloc((size_t)ROWS * HH * 2);
  __bf16* ctxB  = (__bf16*)alloc((size_t)ROWS * HH * 2);
  float*  tmpF  = (float*)alloc((size_t)ROWS * HH * 4);
  __bf16* f1B   = (__bf16*)alloc((size_t)ROWS * FFD * 2);
  __bf16* hmB   = (__bf16*)alloc((size_t)MROWS * HH * 2);
  float*  xmF   = (float*)alloc((size_t)MROWS * EE * 4);
  __bf16* xmB   = (__bf16*)alloc((size_t)MROWS * EE * 2);
  float*  scoresF = (float*)alloc((size_t)MROWS * VV * 4);
  int*   exclB   = (int*)alloc((size_t)ROWS * 4);
  int*   slotSrc = (int*)alloc((size_t)MROWS * 4);
  float* statM   = (float*)alloc((size_t)MROWS * 4);
  float* statS   = (float*)alloc((size_t)MROWS * 4);
  int*   statA   = (int*)alloc((size_t)MROWS * 4);

  auto cvtT = [&](const float* s, __bf16* d, int Kd, int N) {
    dim3 g((N + 31) / 32, (Kd + 31) / 32);
    cvt_transpose_kernel<<<g, 256, 0, stream>>>(s, d, Kd, N);
  };
  auto gemm = [&](const __bf16* A, const __bf16* Bt, const float* bias,
                  float* Cf, __bf16* Cb, int M, int N, int Kd, int act) {
    dim3 g((N + 63) / 64, M / 128);
    gemm_bf16_kernel<<<g, 256, 0, stream>>>(A, Bt, bias, Cf, Cb, M, N, Kd, act);
  };

  // ---- weight conversion (fp32 -> transposed bf16; deterministic) ----
  cvtT(proj_w, projb, EE, HH);
  cvtT(wq, wqb, HH, HH);
  cvtT(wk, wkb, HH, HH);
  cvtT(wv, wvb, HH, HH);
  cvtT(wo, wob, HH, HH);
  cvtT(w1, w1b, HH, FFD);
  cvtT(w2, w2b, FFD, HH);
  cvtT(mlm_w, mlmwb, HH, EE);
  // decoder weight transposed == word_emb [V][E]: plain convert
  cvt_bf16_kernel<<<((long)VV * EE + 255) / 256, 256, 0, stream>>>(
      word_emb, wdecb, (long)VV * EE);

  // ---- embeddings + projection ----
  embed_ln_kernel<<<ROWS, 128, 0, stream>>>(word_emb, pos_emb, tt_emb, ids, tti,
                                            emb_ln_g, emb_ln_b, embb);
  gemm(embb, projb, proj_bi, hF, hB, ROWS, HH, EE, 0);

  // ---- 12 shared ALBERT layers ----
  for (int l = 0; l < NLAYER; ++l) {
    gemm(hB, wqb, bq, nullptr, qB, ROWS, HH, HH, 0);
    gemm(hB, wkb, bk, nullptr, kB, ROWS, HH, HH, 0);
    gemm(hB, wvb, bv, nullptr, vB, ROWS, HH, HH, 0);
    attention_kernel<<<dim3(NHH, BB), 256, 0, stream>>>(qB, kB, vB, am, ctxB);
    gemm(ctxB, wob, bo, tmpF, nullptr, ROWS, HH, HH, 0);
    add_ln_kernel<<<ROWS, 256, 0, stream>>>(hF, tmpF, ln1_g, ln1_b, hF, hB, HH);
    gemm(hB, w1b, b1, nullptr, f1B, ROWS, FFD, HH, 1);          // gelu fused
    gemm(f1B, w2b, b2, tmpF, nullptr, ROWS, HH, FFD, 0);
    add_ln_kernel<<<ROWS, 256, 0, stream>>>(hF, tmpF, ln2_g, ln2_b, hF, hB, HH);
  }

  // ---- MLM head on masked rows only ----
  scan_slots_kernel<<<1, 64, 0, stream>>>(mam, exclB, slotSrc);
  gather_rows_kernel<<<MROWS, 256, 0, stream>>>(hB, slotSrc, hmB);
  gemm(hmB, mlmwb, mlm_b, xmF, nullptr, MROWS, EE, HH, 1);      // gelu fused
  add_ln_kernel<<<MROWS, 256, 0, stream>>>(xmF, nullptr, mlm_ln_g, mlm_ln_b,
                                           nullptr, xmB, EE);
  gemm(xmB, wdecb, dec_b, scoresF, nullptr, MROWS, VV, EE, 0);  // ragged N

  // ---- gumbel softmax (hard, straight-through) + scatter to dense output ----
  gumbel_stats_kernel<<<MROWS, 256, 0, stream>>>(scoresF, rnd, gum, slotSrc,
                                                 statM, statS, statA);
  scatter_out_kernel<<<dim3((VV + 255) / 256, ROWS), 256, 0, stream>>>(
      mam, exclB, ids, slotSrc, scoresF, rnd, gum, statM, statS, statA,
      (float*)d_out);
}
